// HGT_89919435309315
// MI455X (gfx1250) — compile-verified, matrix operand order checked by reading, not run on hardware
//
#include <hip/hip_runtime.h>
#include <math.h>

#define Nn   50000
#define Ee   600000
#define HIDd 128
#define Hh   4
#define Dd   32
#define Rr   4
#define Ll   2
#define Mm   10000
#define MTt  (Nn / 16)            // 3125 row tiles

typedef __bf16 bf16_t;
typedef __attribute__((ext_vector_type(16))) __bf16 bf16x16;
typedef __attribute__((ext_vector_type(8)))  float  v8f;
typedef __attribute__((ext_vector_type(4)))  unsigned int uint32x4;
typedef __attribute__((ext_vector_type(8)))  int          int32x8;
typedef __attribute__((ext_vector_type(4)))  int          int32x4;

#if defined(__has_builtin)
#if __has_builtin(__builtin_amdgcn_tensor_load_to_lds) && __has_builtin(__builtin_amdgcn_s_wait_tensorcnt)
#define USE_TDM 1
#endif
#endif

__device__ __forceinline__ float gelu_exact(float x) {
    return 0.5f * x * (1.0f + erff(x * 0.70710678118654752f));
}

// ---------------------------------------------------------------------------
// TDM: stage one 128x128 f32 matrix from global into LDS with a single DMA.
// D# per CDNA5 ISA ch.8: group0 = {count|flags, lds_addr, global_addr, type},
// group1 = dims/strides (2-D tensor: dim0=dim1=tile0=tile1=128, stride0=128).
// 6-arg builtin form (clang-23 toolchain): groups 2/3 + trailing int32x8 zero.
// ---------------------------------------------------------------------------
#ifdef USE_TDM
__device__ __forceinline__ void tdm_load_w(const float* Wm, unsigned ldsOff) {
    unsigned long long ga = (unsigned long long)(size_t)Wm;
    uint32x4 g0;
    g0[0] = 1u;                                             // count=1
    g0[1] = ldsOff;                                         // lds_addr
    g0[2] = (unsigned)(ga & 0xFFFFFFFFu);                   // global_addr[31:0]
    g0[3] = (unsigned)((ga >> 32) & 0x01FFFFFFu) | (2u << 30); // ga[56:32], type=2
    int32x8 g1;
    g1[0] = (2 << 16);          // data_size = 4 bytes, workgroup_mask = 0
    g1[1] = (128 << 16);        // tensor_dim0[15:0] = 128  (bits 63:48)
    g1[2] = (128 << 16);        // tensor_dim1[15:0] = 128  (bits 95:80)
    g1[3] = (128 << 16);        // tile_dim0 = 128          (bits 127:112)
    g1[4] = 128;                // tile_dim1 = 128, tile_dim2 = 0
    g1[5] = 128;                // tensor_dim0_stride = 128
    g1[6] = 0;                  // tensor_dim1_stride (unused for 2-D)
    g1[7] = 0;
    int32x4 z4 = {0, 0, 0, 0};
    int32x8 z8 = {0, 0, 0, 0, 0, 0, 0, 0};
    __builtin_amdgcn_tensor_load_to_lds(g0, g1, z4, z4, z8, 0);
}
#endif

// ---------------------------------------------------------------------------
// Multi-matrix 128x128 GEMM: C_m = act(A) @ W_m + b_m  (m < nmat)
// Block = 8 waves; wave w owns rows [16*(8*blk+w), +16); per matrix the weight
// is TDM-staged into LDS, then each wave does 8 column tiles x 4 K-steps of
// v_wmma_f32_16x16x32_bf16 with A fragments held in VGPRs across everything.
// FUSE==1: act = exact gelu, epilogue C = gate*C + (1-gate)*resid.
// ---------------------------------------------------------------------------
template <int FUSE>
__global__ __launch_bounds__(256) void hgt_gemm(
    const float* __restrict__ A,
    const float* __restrict__ W0, const float* __restrict__ W1,
    const float* __restrict__ W2,
    const float* __restrict__ b0, const float* __restrict__ b1,
    const float* __restrict__ b2,
    float* __restrict__ C0, float* __restrict__ C1, float* __restrict__ C2,
    int nmat,
    const float* __restrict__ resid, const float* __restrict__ skip_g) {
    __shared__ __align__(16) float wlds[HIDd * HIDd];       // 64 KB staged W

    int wave = threadIdx.x >> 5;
    int lane = threadIdx.x & 31;
    int lo   = lane & 15;
    int hi   = lane >> 4;
    int mt   = blockIdx.x * 8 + wave;
    bool active = (mt < MTt);

    // Preload all 4 K-step A fragments once (ISA 16-bit A layout:
    // lanes 0-15 K = {ka..ka+7, ka+16..ka+23} with ka = ks*32 + 8*hi).
    bf16x16 afrag[4];
    if (active) {
        const float* arow = A + (size_t)((mt << 4) + lo) * HIDd;
#pragma unroll
        for (int ks = 0; ks < 4; ++ks) {
            int ka = ks * 32 + 8 * hi;
#pragma unroll
            for (int e = 0; e < 8; ++e) {
                float a0 = arow[ka + e];
                float a1 = arow[ka + 16 + e];
                if (FUSE) { a0 = gelu_exact(a0); a1 = gelu_exact(a1); }
                afrag[ks][e]     = (bf16_t)a0;
                afrag[ks][e + 8] = (bf16_t)a1;
            }
        }
    }

    float gate = 1.0f, og = 0.0f;
    if (FUSE) {
        float s = skip_g[0];
        gate = 1.0f / (1.0f + __expf(-s));
        og   = 1.0f - gate;
    }

    for (int m = 0; m < nmat; ++m) {
        const float* Wm = (m == 0) ? W0 : (m == 1) ? W1 : W2;
        const float* bm = (m == 0) ? b0 : (m == 1) ? b1 : b2;
        float*       Cm = (m == 0) ? C0 : (m == 1) ? C1 : C2;

        __syncthreads();        // protect LDS reads of previous matrix
#ifdef USE_TDM
        if (wave == 0) {
            tdm_load_w(Wm, (unsigned)(size_t)(void*)wlds);
            __builtin_amdgcn_s_wait_tensorcnt(0);
        }
#else
        for (int i = threadIdx.x; i < HIDd * HIDd / 4; i += 256)
            ((float4*)wlds)[i] = ((const float4*)Wm)[i];
#endif
        __syncthreads();

        if (active) {
#pragma unroll
            for (int nt = 0; nt < 8; ++nt) {
                int c = (nt << 4) + lo;
                v8f acc = {};
#pragma unroll
                for (int ks = 0; ks < 4; ++ks) {
                    int ka = ks * 32 + 8 * hi;
                    bf16x16 b;
#pragma unroll
                    for (int e = 0; e < 8; ++e) {
                        b[e]     = (bf16_t)wlds[(ka + e) * HIDd + c];
                        b[e + 8] = (bf16_t)wlds[(ka + 16 + e) * HIDd + c];
                    }
                    acc = __builtin_amdgcn_wmma_f32_16x16x32_bf16(
                        false, afrag[ks], false, b, (short)0, acc, false, false);
                }
                float bc = bm[c];
#pragma unroll
                for (int vv = 0; vv < 8; ++vv) {
                    int row = (mt << 4) + vv + (hi << 3);   // C/D layout
                    float val = acc[vv] + bc;
                    if (FUSE) val = gate * val + og * resid[(size_t)row * HIDd + c];
                    Cm[(size_t)row * HIDd + c] = val;
                }
            }
        }
    }
}

// ---------------------------------------------------------------------------
// k_all[n,r,h,:] = k[n,h,:] @ rel[r,h]  (D=32 -> exactly one 16x16x32 WMMA)
// ---------------------------------------------------------------------------
__global__ void rel_transform(const float* __restrict__ kv,   // [N,H,D]
                              const float* __restrict__ rel,  // [R,H,D,D]
                              float* __restrict__ outv) {     // [N,R,H,D]
    int t    = blockIdx.x;
    int half = t & 1;
    int rh   = (t >> 1) & 15;
    int mt   = t >> 5;
    int rr   = rh >> 2;
    int hh   = rh & 3;
    int lane = threadIdx.x & 31;
    int lo   = lane & 15;
    int hi   = lane >> 4;
    int node = (mt << 4) + lo;
    int ka   = 8 * hi;
    int c    = half * 16 + lo;

    const float* arow = kv  + ((size_t)node * Hh + hh) * Dd;
    const float* bmat = rel + ((size_t)rr * Hh + hh) * Dd * Dd;
    bf16x16 a, b;
#pragma unroll
    for (int e = 0; e < 8; ++e) {
        a[e]     = (bf16_t)arow[ka + e];
        a[e + 8] = (bf16_t)arow[ka + 16 + e];
        b[e]     = (bf16_t)bmat[(size_t)(ka + e) * Dd + c];
        b[e + 8] = (bf16_t)bmat[(size_t)(ka + 16 + e) * Dd + c];
    }
    v8f acc = {};
    acc = __builtin_amdgcn_wmma_f32_16x16x32_bf16(
        false, a, false, b, (short)0, acc, false, false);
#pragma unroll
    for (int vv = 0; vv < 8; ++vv) {
        int row = (mt << 4) + vv + (hi << 3);
        outv[(((size_t)row * Rr + rr) * Hh + hh) * Dd + c] = acc[vv];
    }
}

// ---------------------------------------------------------------------------
// BatchNorm (training-mode batch statistics)
// ---------------------------------------------------------------------------
__global__ void bn_stats(const float* __restrict__ h, float* __restrict__ bnst) {
    int c      = threadIdx.x;
    int row0   = blockIdx.x * blockDim.y + threadIdx.y;
    int stride = gridDim.x * blockDim.y;
    float s = 0.f, s2 = 0.f;
    for (int r = row0; r < Nn; r += stride) {
        float v = h[(size_t)r * HIDd + c];
        s += v; s2 += v * v;
    }
    atomicAdd(&bnst[c], s);
    atomicAdd(&bnst[HIDd + c], s2);
}

__global__ void bn_apply(float* __restrict__ h, const float* __restrict__ bnst,
                         const float* __restrict__ gamma,
                         const float* __restrict__ beta) {
    long long t = (long long)blockIdx.x * blockDim.x + threadIdx.x;
    if (t >= (long long)Nn * HIDd) return;
    int c = (int)(t & (HIDd - 1));
    float mu  = bnst[c] * (1.0f / (float)Nn);
    float var = bnst[HIDd + c] * (1.0f / (float)Nn) - mu * mu;
    h[t] = (h[t] - mu) * rsqrtf(var + 1e-5f) * gamma[c] + beta[c];
}

// ---------------------------------------------------------------------------
// Edge phase: score -> segment max -> exp/sum -> weighted scatter
// ---------------------------------------------------------------------------
__device__ __forceinline__ unsigned enc_f32(float f) {
    unsigned u = __float_as_uint(f);
    return (u & 0x80000000u) ? ~u : (u | 0x80000000u);
}
__device__ __forceinline__ float dec_f32(unsigned e) {
    unsigned u = (e & 0x80000000u) ? (e & 0x7FFFFFFFu) : ~e;
    return __uint_as_float(u);
}

__global__ void edge_score(const float* __restrict__ q,
                           const float* __restrict__ k_all,
                           const int* __restrict__ ei,
                           const int* __restrict__ etype,
                           const float* __restrict__ p_rel,
                           float* __restrict__ score,
                           unsigned* __restrict__ smax) {
    int t = blockIdx.x * blockDim.x + threadIdx.x;
    if (t >= Ee * Hh) return;
    int e = t >> 2, hh = t & 3;
    int src = ei[e], dst = ei[Ee + e], et = etype[e];
    const float* qr = q     + ((size_t)dst * Hh + hh) * Dd;
    const float* kr = k_all + (((size_t)src * Rr + et) * Hh + hh) * Dd;
    float s = 0.f;
#pragma unroll
    for (int d = 0; d < Dd; d += 4) {
        float4 qa = *(const float4*)(qr + d);
        float4 kb = *(const float4*)(kr + d);
        s += qa.x * kb.x + qa.y * kb.y + qa.z * kb.z + qa.w * kb.w;
    }
    s *= p_rel[et * Hh + hh] * 0.17677669529663687f;  // 1/sqrt(32)
    score[t] = s;
    atomicMax(&smax[((size_t)dst * Rr + et) * Hh + hh], enc_f32(s));
}

__global__ void edge_exp(const float* __restrict__ score,
                         const unsigned* __restrict__ smax,
                         const int* __restrict__ ei,
                         const int* __restrict__ etype,
                         float* __restrict__ expv,
                         float* __restrict__ denom) {
    int t = blockIdx.x * blockDim.x + threadIdx.x;
    if (t >= Ee * Hh) return;
    int e = t >> 2, hh = t & 3;
    int dst = ei[Ee + e], et = etype[e];
    size_t seg = ((size_t)dst * Rr + et) * Hh + hh;
    float ev = __expf(score[t] - dec_f32(smax[seg]));
    expv[t] = ev;
    atomicAdd(&denom[seg], ev);
}

__global__ void edge_agg(const float* __restrict__ expv,
                         const float* __restrict__ denom,
                         const float* __restrict__ v_all,
                         const int* __restrict__ ei,
                         const int* __restrict__ etype,
                         float* __restrict__ attn) {
    long long t = (long long)blockIdx.x * blockDim.x + threadIdx.x;
    if (t >= (long long)Ee * HIDd) return;
    int e = (int)(t >> 7);
    int j = (int)(t & 127);
    int hh = j >> 5, d = j & 31;
    int src = ei[e], dst = ei[Ee + e], et = etype[e];
    float alpha = expv[(size_t)e * Hh + hh] /
                  denom[((size_t)dst * Rr + et) * Hh + hh];
    float val = alpha * v_all[(((size_t)src * Rr + et) * Hh + hh) * Dd + d];
    atomicAdd(&attn[(size_t)dst * HIDd + j], val);
}

// ---------------------------------------------------------------------------
__global__ void zero_u32(unsigned* __restrict__ p, long long n) {
    long long t = (long long)blockIdx.x * blockDim.x + threadIdx.x;
    if (t < n) p[t] = 0u;
}

__global__ void gather_rows(const float* __restrict__ h,
                            const int* __restrict__ idx,
                            float* __restrict__ out) {
    int t = blockIdx.x * blockDim.x + threadIdx.x;
    if (t >= Mm * HIDd) return;
    int row = idx[t >> 7];
    out[t] = h[(size_t)row * HIDd + (t & 127)];
}

// ---------------------------------------------------------------------------
extern "C" void kernel_launch(void* const* d_in, const int* in_sizes, int n_in,
                              void* d_out, int out_size, void* d_ws, size_t ws_size,
                              hipStream_t stream) {
    const float* x      = (const float*)d_in[0];
    const float* proj_w = (const float*)d_in[1];
    const float* proj_b = (const float*)d_in[2];
    const float* bn_g   = (const float*)d_in[3];
    const float* bn_b   = (const float*)d_in[4];
    const float* q_w    = (const float*)d_in[5];
    const float* q_b    = (const float*)d_in[6];
    const float* k_w    = (const float*)d_in[7];
    const float* k_b    = (const float*)d_in[8];
    const float* v_w    = (const float*)d_in[9];
    const float* v_b    = (const float*)d_in[10];
    const float* a_w    = (const float*)d_in[11];
    const float* a_b    = (const float*)d_in[12];
    const float* skip   = (const float*)d_in[13];
    const float* a_rel  = (const float*)d_in[14];
    const float* m_rel  = (const float*)d_in[15];
    const float* p_rel  = (const float*)d_in[16];
    const int*   ei     = (const int*)d_in[17];
    const int*   etype  = (const int*)d_in[18];
    const int*   idx    = (const int*)d_in[19];
    float*       out    = (float*)d_out;

    float* ws = (float*)d_ws;
    size_t off = 0;
    float* h     = ws + off; off += (size_t)Nn * HIDd;
    float* q     = ws + off; off += (size_t)Nn * HIDd;
    float* k     = ws + off; off += (size_t)Nn * HIDd;
    float* v     = ws + off; off += (size_t)Nn * HIDd;
    float* attn  = ws + off; off += (size_t)Nn * HIDd;
    float* k_all = ws + off; off += (size_t)Nn * Rr * Hh * Dd;
    float* v_all = ws + off; off += (size_t)Nn * Rr * Hh * Dd;
    float* score = ws + off; off += (size_t)Ee * Hh;
    float* expv  = ws + off; off += (size_t)Ee * Hh;
    unsigned* smax = (unsigned*)(ws + off); off += (size_t)Nn * Rr * Hh;
    float* denom = ws + off; off += (size_t)Nn * Rr * Hh;
    float* bnst  = ws + off; off += 2 * HIDd;

    const int TPB = 256;
    const unsigned gemmBlocks = (MTt + 7) / 8;            // 391
    const long long nh   = (long long)Nn * HIDd;
    const long long nseg = (long long)Nn * Rr * Hh;
    const long long neh  = (long long)Ee * Hh;
    const long long nes  = (long long)Ee * HIDd;

    // 1) proj + BatchNorm
    hgt_gemm<0><<<gemmBlocks, TPB, 0, stream>>>(
        x, proj_w, proj_w, proj_w, proj_b, proj_b, proj_b,
        h, h, h, 1, nullptr, nullptr);
    zero_u32<<<1, TPB, 0, stream>>>((unsigned*)bnst, 2 * HIDd);
    bn_stats<<<dim3(256), dim3(128, 2), 0, stream>>>(h, bnst);
    bn_apply<<<(unsigned)((nh + TPB - 1) / TPB), TPB, 0, stream>>>(h, bnst, bn_g, bn_b);

    // 2) HGTConv layers
    for (int l = 0; l < Ll; ++l) {
        const float* qwl = q_w + (size_t)l * HIDd * HIDd;
        const float* kwl = k_w + (size_t)l * HIDd * HIDd;
        const float* vwl = v_w + (size_t)l * HIDd * HIDd;
        const float* awl = a_w + (size_t)l * HIDd * HIDd;
        const float* qbl = q_b + (size_t)l * HIDd;
        const float* kbl = k_b + (size_t)l * HIDd;
        const float* vbl = v_b + (size_t)l * HIDd;
        const float* abl = a_b + (size_t)l * HIDd;
        const float* arl = a_rel + (size_t)l * Rr * Hh * Dd * Dd;
        const float* mrl = m_rel + (size_t)l * Rr * Hh * Dd * Dd;
        const float* prl = p_rel + (size_t)l * Rr * Hh;

        // fused q/k/v: h read once, three TDM-staged weight matrices
        hgt_gemm<0><<<gemmBlocks, TPB, 0, stream>>>(
            h, qwl, kwl, vwl, qbl, kbl, vbl, q, k, v, 3, nullptr, nullptr);

        rel_transform<<<dim3(MTt * Rr * Hh * 2), dim3(32), 0, stream>>>(k, arl, k_all);
        rel_transform<<<dim3(MTt * Rr * Hh * 2), dim3(32), 0, stream>>>(v, mrl, v_all);

        zero_u32<<<(unsigned)((nseg + TPB - 1) / TPB), TPB, 0, stream>>>(smax, nseg);
        zero_u32<<<(unsigned)((nseg + TPB - 1) / TPB), TPB, 0, stream>>>((unsigned*)denom, nseg);
        zero_u32<<<(unsigned)((nh + TPB - 1) / TPB), TPB, 0, stream>>>((unsigned*)attn, nh);

        edge_score<<<(unsigned)((neh + TPB - 1) / TPB), TPB, 0, stream>>>(
            q, k_all, ei, etype, prl, score, smax);
        edge_exp<<<(unsigned)((neh + TPB - 1) / TPB), TPB, 0, stream>>>(
            score, smax, ei, etype, expv, denom);
        edge_agg<<<(unsigned)((nes + TPB - 1) / TPB), TPB, 0, stream>>>(
            expv, denom, v_all, ei, etype, attn);

        // out-proj: fused exact-GELU input + sigmoid skip mix, written to h
        hgt_gemm<1><<<gemmBlocks, TPB, 0, stream>>>(
            attn, awl, awl, awl, abl, abl, abl, h, h, h, 1, h, skip + l);
    }

    // 3) final gather
    gather_rows<<<(Mm * HIDd + TPB - 1) / TPB, TPB, 0, stream>>>(h, idx, out);
}
